// TriLinear_Self_Attn_17360257810684
// MI455X (gfx1250) — matrix operand order repaired and checked by reading, not compiled, not run on hardware
//
#include <hip/hip_runtime.h>
#include <hip/hip_bf16.h>

// Problem dims (fixed by the reference)
#define BB 16
#define TT 1024
#define DD 1024
#define D3 3072
#define NEG_INF_F (-1.0e10f)

typedef __attribute__((ext_vector_type(16))) __bf16 v16bf;
typedef __attribute__((ext_vector_type(8)))  float  v8f;

union FragB {
    v16bf v;
    uint4 u[2];
};

__device__ __forceinline__ v8f wmma_bf16(const v16bf& a, const v16bf& b, const v8f& c) {
    return __builtin_amdgcn_wmma_f32_16x16x32_bf16(false, a, false, b, (short)0, c, false, false);
}

// ---------------------------------------------------------------------------
// Kernel 1: per-row prep. qf = row . qw, kf = row . kw, A = bf16(row * dw),
// K = bf16(row).  One block per (b,t) row.
// ---------------------------------------------------------------------------
__global__ void __launch_bounds__(256) prep_kernel(
    const float* __restrict__ mo, const float* __restrict__ qw,
    const float* __restrict__ kw, const float* __restrict__ dw,
    __bf16* __restrict__ Ab, __bf16* __restrict__ Kb,
    float* __restrict__ qf, float* __restrict__ kf)
{
    const int row = blockIdx.x;                 // b*T + t
    const float* src = mo + (size_t)row * DD;
    float accq = 0.f, acck = 0.f;
    for (int d = threadIdx.x; d < DD; d += 256) {
        float v = src[d];
        accq += v * qw[d];
        acck += v * kw[d];
        Kb[(size_t)row * DD + d] = (__bf16)v;
        Ab[(size_t)row * DD + d] = (__bf16)(v * dw[d]);
    }
    __shared__ float sq[256], sk[256];
    sq[threadIdx.x] = accq;
    sk[threadIdx.x] = acck;
    __syncthreads();
    for (int off = 128; off > 0; off >>= 1) {
        if ((int)threadIdx.x < off) {
            sq[threadIdx.x] += sq[threadIdx.x + off];
            sk[threadIdx.x] += sk[threadIdx.x + off];
        }
        __syncthreads();
    }
    if (threadIdx.x == 0) { qf[row] = sq[0]; kf[row] = sk[0]; }
}

// ---------------------------------------------------------------------------
// Kernel 2: tiled transpose  K[b][t][d] -> KT[b][d][t]  (bf16)
// ---------------------------------------------------------------------------
__global__ void __launch_bounds__(256) transpose_kernel(
    const __bf16* __restrict__ Kb, __bf16* __restrict__ KT)
{
    __shared__ __bf16 tile[32][33];
    const int b  = blockIdx.z;
    const int t0 = blockIdx.x * 32;
    const int d0 = blockIdx.y * 32;
    for (int i = threadIdx.y; i < 32; i += 8)
        tile[i][threadIdx.x] = Kb[((size_t)b * TT + t0 + i) * DD + d0 + threadIdx.x];
    __syncthreads();
    for (int i = threadIdx.y; i < 32; i += 8)
        KT[((size_t)b * DD + d0 + i) * TT + t0 + threadIdx.x] = tile[threadIdx.x][i];
}

// ---------------------------------------------------------------------------
// Kernel 3: W[3D][D] fp32 -> WT[D][3D] bf16
// ---------------------------------------------------------------------------
__global__ void __launch_bounds__(256) convw_kernel(
    const float* __restrict__ W, __bf16* __restrict__ WT)
{
    const size_t idx = (size_t)blockIdx.x * 256 + threadIdx.x;  // over 3072*1024
    const int m = (int)(idx / DD);
    const int j = (int)(idx % DD);
    WT[(size_t)j * D3 + m] = (__bf16)W[idx];
}

// ---------------------------------------------------------------------------
// Kernel 4: masked-softmax scores, FlashAttention-2 two-phase (stats then
// recompute+store).  One wave = one 16-row strip; 2 independent B streams
// (32 score columns) per step for load/WMMA overlap.
// ---------------------------------------------------------------------------
__global__ void __launch_bounds__(256) scores_kernel(
    const __bf16* __restrict__ Ab, const __bf16* __restrict__ Kb,
    const float* __restrict__ qf, const float* __restrict__ kf,
    const int* __restrict__ cmask, __bf16* __restrict__ P)
{
    const int lane  = threadIdx.x & 31;
    const int wave  = threadIdx.x >> 5;
    const int strip = blockIdx.x * 8 + wave;    // 0 .. B*T/16-1 (=1023)
    const int b     = strip >> 6;
    const int t0    = (strip & 63) << 4;
    const int half  = lane >> 4;                // 0 or 1
    const int ln    = lane & 15;

    // A-frag base for this lane: row t0+ln, K-chunks offset half*8
    const __bf16* Ap = Ab + ((size_t)b * TT + t0 + ln) * DD + half * 8;
    const __bf16* Kbase = Kb + (size_t)b * TT * DD;

    float qv[8];
#pragma unroll
    for (int r = 0; r < 8; r++)
        qv[r] = qf[b * TT + t0 + 8 * half + r];   // row = r + 8*half

    float m[8], l[8], rl[8];
#pragma unroll
    for (int r = 0; r < 8; r++) { m[r] = -3.0e38f; l[r] = 0.f; rl[r] = 0.f; }

    for (int pass = 0; pass < 2; ++pass) {
        if (pass) {
#pragma unroll
            for (int r = 0; r < 8; r++) rl[r] = 1.0f / l[r];
        }
        for (int s0 = 0; s0 < TT; s0 += 32) {
            // Two column tiles: s0 and s0+16.  B-frag lane column = sX + ln,
            // K-range d = k + half*8 + {0..7, 16..23} (matches A-frag order).
            const __bf16* Bp0 = Kbase + (size_t)(s0 + ln) * DD + half * 8;
            const __bf16* Bp1 = Bp0 + (size_t)16 * DD;
            v8f acc0 = {}, acc1 = {};
#pragma unroll
            for (int k = 0; k < DD; k += 32) {
                FragB a, b0, b1;
                a.u[0]  = *(const uint4*)(Ap + k);
                a.u[1]  = *(const uint4*)(Ap + k + 16);
                b0.u[0] = *(const uint4*)(Bp0 + k);
                b0.u[1] = *(const uint4*)(Bp0 + k + 16);
                b1.u[0] = *(const uint4*)(Bp1 + k);
                b1.u[1] = *(const uint4*)(Bp1 + k + 16);
                acc0 = wmma_bf16(a.v, b0.v, acc0);
                acc1 = wmma_bf16(a.v, b1.v, acc1);
            }
#pragma unroll
            for (int h2 = 0; h2 < 2; ++h2) {
                const v8f&  acc = h2 ? acc1 : acc0;
                const int   sc  = s0 + 16 * h2 + ln;     // this lane's column
                const float kfv = kf[b * TT + sc];
                const bool  cm_s = cmask[b * TT + sc] != 0;
                float vv[8];
#pragma unroll
                for (int r = 0; r < 8; r++) {
                    const int trow = t0 + r + 8 * half;
                    const bool masked = cm_s || (trow == sc);
                    vv[r] = masked ? NEG_INF_F : (acc[r] + qv[r] + kfv);
                }
                if (pass == 0) {
#pragma unroll
                    for (int r = 0; r < 8; r++) {
                        float tmax = vv[r];
                        tmax = fmaxf(tmax, __shfl_xor(tmax, 1, 16));
                        tmax = fmaxf(tmax, __shfl_xor(tmax, 2, 16));
                        tmax = fmaxf(tmax, __shfl_xor(tmax, 4, 16));
                        tmax = fmaxf(tmax, __shfl_xor(tmax, 8, 16));
                        const float mnew = fmaxf(m[r], tmax);
                        float ts = __expf(vv[r] - mnew);
                        ts += __shfl_xor(ts, 1, 16);
                        ts += __shfl_xor(ts, 2, 16);
                        ts += __shfl_xor(ts, 4, 16);
                        ts += __shfl_xor(ts, 8, 16);
                        l[r] = l[r] * __expf(m[r] - mnew) + ts;
                        m[r] = mnew;
                    }
                } else {
#pragma unroll
                    for (int r = 0; r < 8; r++) {
                        const float p = __expf(vv[r] - m[r]) * rl[r];
                        P[((size_t)b * TT + t0 + r + 8 * half) * TT + sc] = (__bf16)p;
                    }
                }
            }
        }
    }
}

// ---------------------------------------------------------------------------
// Kernel 5: SA = P @ K  (per batch [T,T]x[T,D]), bf16 out.
// One wave = 16x64 output (4 independent B streams / accumulators).
// ---------------------------------------------------------------------------
__global__ void __launch_bounds__(256) pv_kernel(
    const __bf16* __restrict__ P, const __bf16* __restrict__ KT,
    __bf16* __restrict__ SA)
{
    const int lane = threadIdx.x & 31;
    const int wave = threadIdx.x >> 5;
    const int tile = blockIdx.x * 8 + wave;     // over (B*T/16) * (D/64) = 16384
    const int rt   = tile >> 4;                 // flattened row tile (0..1023)
    const int g    = tile & 15;                 // d-group of 64 cols
    const int b    = rt >> 6;
    const int t0   = (rt & 63) << 4;
    const int d0   = g << 6;
    const int half = lane >> 4;
    const int ln   = lane & 15;

    const __bf16* Ap = P + ((size_t)b * TT + t0 + ln) * TT + half * 8;
    const __bf16* Bp[4];
#pragma unroll
    for (int q = 0; q < 4; q++)
        Bp[q] = KT + ((size_t)b * DD + d0 + 16 * q + ln) * TT + half * 8;

    v8f acc[4] = {};
#pragma unroll 4
    for (int s = 0; s < TT; s += 32) {
        FragB a;
        a.u[0] = *(const uint4*)(Ap + s);
        a.u[1] = *(const uint4*)(Ap + s + 16);
#pragma unroll
        for (int q = 0; q < 4; q++) {
            FragB bf;
            bf.u[0] = *(const uint4*)(Bp[q] + s);
            bf.u[1] = *(const uint4*)(Bp[q] + s + 16);
            acc[q] = wmma_bf16(a.v, bf.v, acc[q]);
        }
    }
#pragma unroll
    for (int q = 0; q < 4; q++)
#pragma unroll
        for (int r = 0; r < 8; r++)
            SA[((size_t)b * TT + t0 + r + 8 * half) * DD + d0 + 16 * q + ln] =
                (__bf16)acc[q][r];
}

// ---------------------------------------------------------------------------
// Kernel 6: out = relu([SA | K | SA*K] @ W + bias) + residual, row-masked.
// One wave = 16x32 fp32 output (2 independent B streams).
// ---------------------------------------------------------------------------
__global__ void __launch_bounds__(256) out_kernel(
    const __bf16* __restrict__ SA, const __bf16* __restrict__ Kb,
    const __bf16* __restrict__ WT, const float* __restrict__ bias,
    const float* __restrict__ resid, const int* __restrict__ cmask,
    float* __restrict__ out)
{
    const int lane = threadIdx.x & 31;
    const int wave = threadIdx.x >> 5;
    const int tile = blockIdx.x * 8 + wave;     // over (B*T/16) * (D/32) = 32768
    const int rt   = tile >> 5;
    const int g    = tile & 31;                 // j-group of 32 cols
    const int r0   = rt << 4;                   // flattened B*T row
    const int j0   = g << 5;
    const int half = lane >> 4;
    const int ln   = lane & 15;

    const __bf16* SArow = SA + (size_t)(r0 + ln) * DD + half * 8;
    const __bf16* Krow  = Kb + (size_t)(r0 + ln) * DD + half * 8;
    const __bf16* Bbase0 = WT + (size_t)(j0 + ln) * D3 + half * 8;
    const __bf16* Bbase1 = Bbase0 + (size_t)16 * D3;

    v8f acc0 = {}, acc1 = {};
    for (int seg = 0; seg < 3; ++seg) {
        const __bf16* Bs0 = Bbase0 + seg * DD;
        const __bf16* Bs1 = Bbase1 + seg * DD;
#pragma unroll 8
        for (int k = 0; k < DD; k += 32) {
            FragB a;
            if (seg == 0) {
                a.u[0] = *(const uint4*)(SArow + k);
                a.u[1] = *(const uint4*)(SArow + k + 16);
            } else if (seg == 1) {
                a.u[0] = *(const uint4*)(Krow + k);
                a.u[1] = *(const uint4*)(Krow + k + 16);
            } else {
                FragB fs, fk;
                fs.u[0] = *(const uint4*)(SArow + k);
                fs.u[1] = *(const uint4*)(SArow + k + 16);
                fk.u[0] = *(const uint4*)(Krow + k);
                fk.u[1] = *(const uint4*)(Krow + k + 16);
#pragma unroll
                for (int i = 0; i < 16; i++)
                    a.v[i] = (__bf16)((float)fs.v[i] * (float)fk.v[i]);
            }
            FragB b0, b1;
            b0.u[0] = *(const uint4*)(Bs0 + k);
            b0.u[1] = *(const uint4*)(Bs0 + k + 16);
            b1.u[0] = *(const uint4*)(Bs1 + k);
            b1.u[1] = *(const uint4*)(Bs1 + k + 16);
            acc0 = wmma_bf16(a.v, b0.v, acc0);
            acc1 = wmma_bf16(a.v, b1.v, acc1);
        }
    }

    const float bj0 = bias[j0 + ln];
    const float bj1 = bias[j0 + 16 + ln];
#pragma unroll
    for (int r = 0; r < 8; r++) {
        const int row = r0 + r + 8 * half;
        const bool cm = cmask[row] != 0;
        float v0 = fmaxf(acc0[r] + bj0, 0.0f) + resid[(size_t)row * DD + j0 + ln];
        float v1 = fmaxf(acc1[r] + bj1, 0.0f) + resid[(size_t)row * DD + j0 + 16 + ln];
        out[(size_t)row * DD + j0 + ln]      = cm ? 0.0f : v0;
        out[(size_t)row * DD + j0 + 16 + ln] = cm ? 0.0f : v1;
    }
}

// ---------------------------------------------------------------------------
extern "C" void kernel_launch(void* const* d_in, const int* in_sizes, int n_in,
                              void* d_out, int out_size, void* d_ws, size_t ws_size,
                              hipStream_t stream) {
    const float* mo    = (const float*)d_in[0];   // [B,T,D]
    const int*   cmask = (const int*)d_in[1];     // [B,T]
    const float* qw    = (const float*)d_in[2];   // [D,1]
    const float* kw    = (const float*)d_in[3];   // [D,1]
    const float* dw    = (const float*)d_in[4];   // [D]
    const float* W     = (const float*)d_in[5];   // [3D,D]
    const float* bias  = (const float*)d_in[6];   // [D]
    float* out = (float*)d_out;

    // Workspace layout (bf16 unless noted)
    char* w = (char*)d_ws;
    const size_t szBTD = (size_t)BB * TT * DD * sizeof(__bf16);   // 32 MB
    const size_t szBTT = (size_t)BB * TT * TT * sizeof(__bf16);   // 32 MB
    __bf16* Ab = (__bf16*)w;           w += szBTD;   // q * dot_w
    __bf16* Kb = (__bf16*)w;           w += szBTD;   // k
    __bf16* KT = (__bf16*)w;           w += szBTD;   // k transposed [b][d][t]
    __bf16* P  = (__bf16*)w;           w += szBTT;   // softmax probs
    __bf16* SA = (__bf16*)w;           w += szBTD;   // self_attn
    __bf16* WT = (__bf16*)w;           w += (size_t)DD * D3 * sizeof(__bf16); // 6 MB
    float*  qf = (float*)w;            w += (size_t)BB * TT * sizeof(float);
    float*  kf = (float*)w;            w += (size_t)BB * TT * sizeof(float);

    prep_kernel<<<BB * TT, 256, 0, stream>>>(mo, qw, kw, dw, Ab, Kb, qf, kf);
    transpose_kernel<<<dim3(TT / 32, DD / 32, BB), dim3(32, 8, 1), 0, stream>>>(Kb, KT);
    convw_kernel<<<(D3 * DD) / 256, 256, 0, stream>>>(W, WT);
    scores_kernel<<<(BB * TT) / 128, 256, 0, stream>>>(Ab, Kb, qf, kf, cmask, P);
    pv_kernel<<<((BB * TT / 16) * (DD / 64)) / 8, 256, 0, stream>>>(P, KT, SA);
    out_kernel<<<((BB * TT / 16) * (DD / 32)) / 8, 256, 0, stream>>>(SA, Kb, WT, bias, mo, cmask, out);
}
